// SoftMoE_69140383531416
// MI455X (gfx1250) — matrix operand order repaired and checked by reading, not compiled, not run on hardware
//
#include <hip/hip_runtime.h>

#define BB 4
#define LL 4096
#define DM 1024
#define NH 16
#define DH 64
#define SS 128
#define EE 8
#define PP 16
#define DFF 4096

typedef __attribute__((ext_vector_type(16))) __bf16 bf16x16;
typedef __attribute__((ext_vector_type(8)))  float  floatx8;

__device__ __forceinline__ floatx8 wmma_bf16(bf16x16 a, bf16x16 b, floatx8 c) {
  return __builtin_amdgcn_wmma_f32_16x16x32_bf16(false, a, false, b, (short)0, c,
                                                 false, false);
}

// A fragment: 16(M) x 32(K) tile from row-major LDS [rows][ld].
// Lane l holds M = l&15; K base = k0 + 16*(l>=16); elem j -> K += j.
__device__ __forceinline__ bf16x16 load_a_frag(const __bf16* s, int ld, int m0,
                                               int k0, int lane) {
  const __bf16* p = s + (m0 + (lane & 15)) * ld + k0 + ((lane >> 4) << 4);
  bf16x16 r;
#pragma unroll
  for (int j = 0; j < 16; ++j) r[j] = p[j];
  return r;
}

// B fragment where the LDS tile is stored [N][K] row-major (B = tile^T):
// lane l holds N = l&15; elem j -> K = k0 + 16*(l>=16) + j.
__device__ __forceinline__ bf16x16 load_b_fragT(const __bf16* s, int ld, int n0,
                                                int k0, int lane) {
  const __bf16* p = s + (n0 + (lane & 15)) * ld + k0 + ((lane >> 4) << 4);
  bf16x16 r;
#pragma unroll
  for (int j = 0; j < 16; ++j) r[j] = p[j];
  return r;
}

// B fragment where the LDS tile is stored [K][N] row-major.
__device__ __forceinline__ bf16x16 load_b_fragN(const __bf16* s, int ld, int n0,
                                                int k0, int lane) {
  int n = n0 + (lane & 15);
  int kb = k0 + ((lane >> 4) << 4);
  bf16x16 r;
#pragma unroll
  for (int j = 0; j < 16; ++j) r[j] = s[(kb + j) * ld + n];
  return r;
}

__device__ __forceinline__ void cvt_store8(__bf16* d, float4 a, float4 b) {
  d[0] = (__bf16)a.x; d[1] = (__bf16)a.y; d[2] = (__bf16)a.z; d[3] = (__bf16)a.w;
  d[4] = (__bf16)b.x; d[5] = (__bf16)b.y; d[6] = (__bf16)b.z; d[7] = (__bf16)b.w;
}

// ---------------------------------------------------------------------------
// Generic GEMM: C[M,N] = A[M,K] * W[N,K]^T + bias[N]   (fp32 in/out, bf16 WMMA)
// Block tile 64(M) x 128(N), 8 waves = 4 m-tiles x 2 n-halves, K step 32.
// Software-pipelined: next k-slab global loads issued before current WMMAs.
// ---------------------------------------------------------------------------
__global__ __launch_bounds__(256, 1)
void gemm_bias_kernel(const float* __restrict__ A, const float* __restrict__ W,
                      const float* __restrict__ bias, float* __restrict__ C,
                      int M, int N, int K) {
  __shared__ __bf16 sA[64 * 32];
  __shared__ __bf16 sW[128 * 32];
  const int tid = threadIdx.x, wave = tid >> 5, lane = tid & 31;
  const int mt = wave & 3, nh = wave >> 2;
  const int m0 = blockIdx.y * 64, n0 = blockIdx.x * 128;

  const int rA = tid >> 2, cA = (tid & 3) * 8;   // A: 8 floats/thread
  const int rW = tid >> 1, cW = (tid & 1) * 16;  // W: 16 floats/thread

  float4 pa0, pa1, pw[4];
  auto prefetch = [&](int k0) {
    const float* sa = A + (long)(m0 + rA) * K + k0 + cA;
    pa0 = ((const float4*)sa)[0];
    pa1 = ((const float4*)sa)[1];
    const float* sw = W + (long)(n0 + rW) * K + k0 + cW;
#pragma unroll
    for (int q = 0; q < 4; ++q) pw[q] = ((const float4*)sw)[q];
  };

  floatx8 acc[4];
#pragma unroll
  for (int i = 0; i < 4; ++i) acc[i] = (floatx8)0.0f;

  prefetch(0);
  for (int k0 = 0; k0 < K; k0 += 32) {
    __syncthreads();
    cvt_store8(sA + rA * 32 + cA, pa0, pa1);
    cvt_store8(sW + rW * 32 + cW, pw[0], pw[1]);
    cvt_store8(sW + rW * 32 + cW + 8, pw[2], pw[3]);
    __syncthreads();
    if (k0 + 32 < K) prefetch(k0 + 32);  // overlap with WMMA below

    bf16x16 a = load_a_frag(sA, 32, mt * 16, 0, lane);
    bf16x16 b[4];
#pragma unroll
    for (int nt = 0; nt < 4; ++nt)
      b[nt] = load_b_fragT(sW, 32, nh * 64 + nt * 16, 0, lane);
#pragma unroll
    for (int nt = 0; nt < 4; ++nt) acc[nt] = wmma_bf16(a, b[nt], acc[nt]);
  }
#pragma unroll
  for (int nt = 0; nt < 4; ++nt)
#pragma unroll
    for (int v = 0; v < 8; ++v) {
      int m = mt * 16 + v + ((lane >> 4) << 3);
      int n = n0 + nh * 64 + nt * 16 + (lane & 15);
      C[(long)(m0 + m) * N + n] = acc[nt][v] + bias[n];
    }
}

// ---------------------------------------------------------------------------
// Flash attention with optional split-K over keys (flash-decoding style).
// Block = one (b, h, q-tile of 128, split). 8 waves x 16 q-rows.
// nsplit==1: normalized output to Og.  nsplit>1: unnormalized O + (m,l) to ws.
// ---------------------------------------------------------------------------
__global__ __launch_bounds__(256, 1)
void attn_kernel(const float* __restrict__ Qg, const float* __restrict__ Kg,
                 const float* __restrict__ Vg, float* __restrict__ Og,
                 int Lq, int Lk, long qbs, long kvbs, long obs,
                 int nsplit, float* __restrict__ po, float* __restrict__ pm,
                 float* __restrict__ pl) {
  __shared__ __bf16 sQ[128 * DH];      // 16KB
  __shared__ __bf16 sKP[8 * 16 * 128]; // 32KB: K chunk (first 8192) / P tiles
  __shared__ __bf16 sV[128 * DH];      // 16KB

  const int nqt = Lq >> 7;
  const int sp = blockIdx.x % nsplit;
  const int rest = blockIdx.x / nsplit;
  const int qt = rest % nqt;
  const int bh = rest / nqt;
  const int h = bh % NH, b = bh / NH;
  const float* Qb = Qg + (long)b * qbs + h * DH;
  const float* Kb = Kg + (long)b * kvbs + h * DH;
  const float* Vb = Vg + (long)b * kvbs + h * DH;

  const int seg = Lk / nsplit;
  const int c0lo = sp * seg, c0hi = c0lo + seg;

  const int tid = threadIdx.x, wave = tid >> 5, lane = tid & 31;

  // Stage Q tile (128x64) with 1/sqrt(dh) folded in.
  for (int i = tid; i < 128 * (DH / 4); i += 256) {
    int r = i >> 4, c = (i & 15) << 2;
    float4 f = *(const float4*)(Qb + (long)(qt * 128 + r) * DM + c);
    __bf16* d = sQ + r * DH + c;
    d[0] = (__bf16)(f.x * 0.125f); d[1] = (__bf16)(f.y * 0.125f);
    d[2] = (__bf16)(f.z * 0.125f); d[3] = (__bf16)(f.w * 0.125f);
  }

  float4 pk[8], pv[8];
  auto prefetch_kv = [&](int c0) {
#pragma unroll
    for (int q = 0; q < 8; ++q) {
      int i = tid + q * 256;
      int r = i >> 4, c = (i & 15) << 2;
      long g = (long)(c0 + r) * DM + c;
      pk[q] = *(const float4*)(Kb + g);
      pv[q] = *(const float4*)(Vb + g);
    }
  };

  floatx8 o_acc[4];
#pragma unroll
  for (int i = 0; i < 4; ++i) o_acc[i] = (floatx8)0.0f;
  float m_run[8], l_run[8];
#pragma unroll
  for (int v = 0; v < 8; ++v) { m_run[v] = -1e30f; l_run[v] = 0.0f; }

  __bf16* myP = sKP + wave * 16 * 128;

  prefetch_kv(c0lo);
  for (int c0 = c0lo; c0 < c0hi; c0 += 128) {
    __syncthreads();  // prev-iter consumers of sKP/sV done (also covers sQ)
#pragma unroll
    for (int q = 0; q < 8; ++q) {
      int i = tid + q * 256;
      int r = i >> 4, c = (i & 15) << 2;
      __bf16* dk = sKP + r * DH + c;
      __bf16* dv = sV + r * DH + c;
      dk[0] = (__bf16)pk[q].x; dk[1] = (__bf16)pk[q].y;
      dk[2] = (__bf16)pk[q].z; dk[3] = (__bf16)pk[q].w;
      dv[0] = (__bf16)pv[q].x; dv[1] = (__bf16)pv[q].y;
      dv[2] = (__bf16)pv[q].z; dv[3] = (__bf16)pv[q].w;
    }
    __syncthreads();
    if (c0 + 128 < c0hi) prefetch_kv(c0 + 128);  // overlap with compute below

    // Scores: 16 q-rows x 128 keys = 8 tiles; A frag reused across all tiles.
    floatx8 s[8];
#pragma unroll
    for (int nt = 0; nt < 8; ++nt) s[nt] = (floatx8)0.0f;
#pragma unroll
    for (int kk = 0; kk < 2; ++kk) {
      bf16x16 a = load_a_frag(sQ, DH, wave * 16, kk * 32, lane);
#pragma unroll
      for (int nt = 0; nt < 8; ++nt) {
        bf16x16 bq = load_b_fragT(sKP, DH, nt * 16, kk * 32, lane);
        s[nt] = wmma_bf16(a, bq, s[nt]);
      }
    }
    __syncthreads();  // all waves done reading K before P overwrites it

    // Online softmax (row = (v, lane-group); reductions across 16 lanes).
    float scl[8], rsum[8];
#pragma unroll
    for (int v = 0; v < 8; ++v) {
      float mx = s[0][v];
#pragma unroll
      for (int nt = 1; nt < 8; ++nt) mx = fmaxf(mx, s[nt][v]);
      mx = fmaxf(mx, __shfl_xor(mx, 1));
      mx = fmaxf(mx, __shfl_xor(mx, 2));
      mx = fmaxf(mx, __shfl_xor(mx, 4));
      mx = fmaxf(mx, __shfl_xor(mx, 8));
      float mn = fmaxf(m_run[v], mx);
      scl[v] = __expf(m_run[v] - mn);
      m_run[v] = mn;
      rsum[v] = 0.0f;
    }
#pragma unroll
    for (int nt = 0; nt < 8; ++nt)
#pragma unroll
      for (int v = 0; v < 8; ++v) {
        float p = __expf(s[nt][v] - m_run[v]);
        rsum[v] += p;
        int r = v + ((lane >> 4) << 3);
        myP[r * 128 + nt * 16 + (lane & 15)] = (__bf16)p;
      }
#pragma unroll
    for (int v = 0; v < 8; ++v) {
      float rs = rsum[v];
      rs += __shfl_xor(rs, 1);
      rs += __shfl_xor(rs, 2);
      rs += __shfl_xor(rs, 4);
      rs += __shfl_xor(rs, 8);
      l_run[v] = l_run[v] * scl[v] + rs;
    }
#pragma unroll
    for (int d = 0; d < 4; ++d)
#pragma unroll
      for (int v = 0; v < 8; ++v) o_acc[d][v] *= scl[v];

    // P(16x128) @ V(128x64): A frag per k-step reused across the 4 d-tiles.
#pragma unroll
    for (int kt = 0; kt < 4; ++kt) {
      bf16x16 a = load_a_frag(myP, 128, 0, kt * 32, lane);
#pragma unroll
      for (int d = 0; d < 4; ++d) {
        bf16x16 bv = load_b_fragN(sV, DH, d * 16, kt * 32, lane);
        o_acc[d] = wmma_bf16(a, bv, o_acc[d]);
      }
    }
  }

  if (nsplit == 1) {
    float* Ob = Og + (long)b * obs + h * DH;
#pragma unroll
    for (int d = 0; d < 4; ++d)
#pragma unroll
      for (int v = 0; v < 8; ++v) {
        int m = v + ((lane >> 4) << 3);
        int grow = qt * 128 + wave * 16 + m;
        int col = d * 16 + (lane & 15);
        Ob[(long)grow * DM + col] = o_acc[d][v] / l_run[v];
      }
  } else {
    const long pbase = (long)((bh * nqt + qt) * nsplit + sp);
    float* pob = po + pbase * (128 * DH);
#pragma unroll
    for (int d = 0; d < 4; ++d)
#pragma unroll
      for (int v = 0; v < 8; ++v) {
        int m = v + ((lane >> 4) << 3);
        pob[(wave * 16 + m) * DH + d * 16 + (lane & 15)] = o_acc[d][v];
      }
    if ((lane & 15) == 0) {
#pragma unroll
      for (int v = 0; v < 8; ++v) {
        int r = wave * 16 + v + ((lane >> 4) << 3);
        pm[pbase * 128 + r] = m_run[v];
        pl[pbase * 128 + r] = l_run[v];
      }
    }
  }
}

// Merge split-K attention partials: O = sum_s w_s*O_s / sum_s w_s*l_s.
__global__ __launch_bounds__(256, 1)
void attn_merge_kernel(const float* __restrict__ po, const float* __restrict__ pm,
                       const float* __restrict__ pl, float* __restrict__ Og,
                       int nsplit, long obs) {
  const int bh = blockIdx.x;  // nqt == 1 (dispatch only)
  const int h = bh % NH, b = bh / NH;
  float* Ob = Og + (long)b * obs + h * DH;
  const int tid = threadIdx.x;
  for (int i = tid; i < 128 * DH; i += 256) {
    int r = i >> 6, c = i & 63;
    float mstar = -1e30f;
    for (int s = 0; s < nsplit; ++s)
      mstar = fmaxf(mstar, pm[(long)(bh * nsplit + s) * 128 + r]);
    float lstar = 0.0f, oacc = 0.0f;
    for (int s = 0; s < nsplit; ++s) {
      long pb = (long)(bh * nsplit + s);
      float w = __expf(pm[pb * 128 + r] - mstar);
      lstar += w * pl[pb * 128 + r];
      oacc += w * po[pb * (128 * DH) + r * DH + c];
    }
    Ob[(long)r * DM + c] = oacc / lstar;
  }
}

// ---------------------------------------------------------------------------
// LayerNorm over D=1024 per (b,slot) row; writes bf16 h regrouped per expert:
// h[e][b*16 + p][d], so expert GEMMs see contiguous 64-row A panels.
// ---------------------------------------------------------------------------
__global__ __launch_bounds__(256, 1)
void ln_kernel(const float* __restrict__ o1, const float* __restrict__ scale,
               const float* __restrict__ bias, __bf16* __restrict__ h) {
  const int row = blockIdx.x;          // b*S + s
  const int b = row / SS, s = row % SS;
  const int e = s >> 4, p = s & 15;
  const float* x = o1 + (long)row * DM;
  const int tid = threadIdx.x, wave = tid >> 5, lane = tid & 31;

  float vals[4], lsum = 0.0f, lsq = 0.0f;
#pragma unroll
  for (int i = 0; i < 4; ++i) {
    float v = x[tid + i * 256];
    vals[i] = v; lsum += v; lsq += v * v;
  }
#pragma unroll
  for (int off = 1; off < 32; off <<= 1) {
    lsum += __shfl_xor(lsum, off);
    lsq += __shfl_xor(lsq, off);
  }
  __shared__ float redS[8], redQ[8];
  if (lane == 0) { redS[wave] = lsum; redQ[wave] = lsq; }
  __syncthreads();
  float ts = 0.0f, tq = 0.0f;
#pragma unroll
  for (int w = 0; w < 8; ++w) { ts += redS[w]; tq += redQ[w]; }
  float mu = ts * (1.0f / DM);
  float var = tq * (1.0f / DM) - mu * mu;
  float rstd = rsqrtf(var + 1e-5f);

  __bf16* hr = h + ((long)e * 64 + b * 16 + p) * DM;
#pragma unroll
  for (int i = 0; i < 4; ++i) {
    int d = tid + i * 256;
    float hv = (vals[i] - mu) * rstd * scale[e * DM + d] + bias[e * DM + d];
    hr[d] = (__bf16)hv;
  }
}

// ---------------------------------------------------------------------------
// Fused u/g expert GEMM: u = h*w_in^T + b_in, g = h*w_gate^T + b_gate,
// y = silu(u)*sigmoid(g) -> bf16. Block = (expert, 64 DFF cols), M=64.
// ---------------------------------------------------------------------------
__global__ __launch_bounds__(256, 1)
void mlp_ug_kernel(const __bf16* __restrict__ h, const float* __restrict__ w_in,
                   const float* __restrict__ b_in, const float* __restrict__ w_gate,
                   const float* __restrict__ b_gate, __bf16* __restrict__ y) {
  __shared__ __bf16 sA[64 * 32], sWu[64 * 32], sWg[64 * 32];
  const int e = blockIdx.y, n0 = blockIdx.x * 64;
  const int tid = threadIdx.x, wave = tid >> 5, lane = tid & 31;
  const int mt = wave & 3, np = wave >> 2;  // np: 32-col half
  const __bf16* hb = h + (long)e * 64 * DM;
  const float* wu = w_in + (long)e * DFF * DM;
  const float* wg = w_gate + (long)e * DFF * DM;

  const int r = tid >> 2, c = (tid & 3) * 8;
  uint4 pa;
  float4 pu0, pu1, pg0, pg1;
  auto prefetch = [&](int k0) {
    pa = *(const uint4*)(hb + (long)r * DM + k0 + c);
    const float* s0 = wu + (long)(n0 + r) * DM + k0 + c;
    const float* s1 = wg + (long)(n0 + r) * DM + k0 + c;
    pu0 = ((const float4*)s0)[0]; pu1 = ((const float4*)s0)[1];
    pg0 = ((const float4*)s1)[0]; pg1 = ((const float4*)s1)[1];
  };

  floatx8 au[2], ag[2];
#pragma unroll
  for (int t = 0; t < 2; ++t) { au[t] = (floatx8)0.0f; ag[t] = (floatx8)0.0f; }

  prefetch(0);
  for (int k0 = 0; k0 < DM; k0 += 32) {
    __syncthreads();
    *(uint4*)(sA + r * 32 + c) = pa;
    cvt_store8(sWu + r * 32 + c, pu0, pu1);
    cvt_store8(sWg + r * 32 + c, pg0, pg1);
    __syncthreads();
    if (k0 + 32 < DM) prefetch(k0 + 32);

    bf16x16 a = load_a_frag(sA, 32, mt * 16, 0, lane);
    bf16x16 bu[2], bg[2];
#pragma unroll
    for (int t = 0; t < 2; ++t) {
      bu[t] = load_b_fragT(sWu, 32, np * 32 + t * 16, 0, lane);
      bg[t] = load_b_fragT(sWg, 32, np * 32 + t * 16, 0, lane);
    }
#pragma unroll
    for (int t = 0; t < 2; ++t) {
      au[t] = wmma_bf16(a, bu[t], au[t]);
      ag[t] = wmma_bf16(a, bg[t], ag[t]);
    }
  }
#pragma unroll
  for (int t = 0; t < 2; ++t)
#pragma unroll
    for (int v = 0; v < 8; ++v) {
      int m = mt * 16 + v + ((lane >> 4) << 3);
      int f = n0 + np * 32 + t * 16 + (lane & 15);
      float uu = au[t][v] + b_in[e * DFF + f];
      float gg = ag[t][v] + b_gate[e * DFF + f];
      float yy = (uu / (1.0f + __expf(-uu))) * (1.0f / (1.0f + __expf(-gg)));
      y[((long)e * 64 + m) * DFF + f] = (__bf16)yy;
    }
}

// ---------------------------------------------------------------------------
// Expert output GEMM: o2 = y*w_out^T + b_out + o1 (residual), K=DFF.
// ---------------------------------------------------------------------------
__global__ __launch_bounds__(256, 1)
void mlp_out_kernel(const __bf16* __restrict__ y, const float* __restrict__ w_out,
                    const float* __restrict__ b_out, const float* __restrict__ o1,
                    float* __restrict__ o2) {
  __shared__ __bf16 sA[64 * 32], sW[64 * 32];
  const int e = blockIdx.y, n0 = blockIdx.x * 64;
  const int tid = threadIdx.x, wave = tid >> 5, lane = tid & 31;
  const int mt = wave & 3, nh = wave >> 2;
  const __bf16* yb = y + (long)e * 64 * DFF;
  const float* wb = w_out + (long)e * DM * DFF;

  const int r = tid >> 2, c = (tid & 3) * 8;
  uint4 pa;
  float4 pw0, pw1;
  auto prefetch = [&](int k0) {
    pa = *(const uint4*)(yb + (long)r * DFF + k0 + c);
    const float* src = wb + (long)(n0 + r) * DFF + k0 + c;
    pw0 = ((const float4*)src)[0]; pw1 = ((const float4*)src)[1];
  };

  floatx8 acc[2];
#pragma unroll
  for (int t = 0; t < 2; ++t) acc[t] = (floatx8)0.0f;

  prefetch(0);
  for (int k0 = 0; k0 < DFF; k0 += 32) {
    __syncthreads();
    *(uint4*)(sA + r * 32 + c) = pa;
    cvt_store8(sW + r * 32 + c, pw0, pw1);
    __syncthreads();
    if (k0 + 32 < DFF) prefetch(k0 + 32);

    bf16x16 a = load_a_frag(sA, 32, mt * 16, 0, lane);
    bf16x16 b0 = load_b_fragT(sW, 32, nh * 32 + 0, 0, lane);
    bf16x16 b1 = load_b_fragT(sW, 32, nh * 32 + 16, 0, lane);
    acc[0] = wmma_bf16(a, b0, acc[0]);
    acc[1] = wmma_bf16(a, b1, acc[1]);
  }
#pragma unroll
  for (int t = 0; t < 2; ++t)
#pragma unroll
    for (int v = 0; v < 8; ++v) {
      int m = mt * 16 + v + ((lane >> 4) << 3);      // 0..63 = b*16 + p
      int bb = m >> 4, p = m & 15;
      int d = n0 + nh * 32 + t * 16 + (lane & 15);
      long row = (long)bb * SS + e * 16 + p;
      o2[row * DM + d] = acc[t][v] + b_out[e * DM + d] + o1[row * DM + d];
    }
}

// ---------------------------------------------------------------------------
extern "C" void kernel_launch(void* const* d_in, const int* in_sizes, int n_in,
                              void* d_out, int out_size, void* d_ws, size_t ws_size,
                              hipStream_t stream) {
  const float* x      = (const float*)d_in[0];
  const float* slots  = (const float*)d_in[1];
  const float* wkd    = (const float*)d_in[2];
  const float* bkd    = (const float*)d_in[3];
  const float* wqc    = (const float*)d_in[4];
  const float* bqc    = (const float*)d_in[5];
  const float* wkc    = (const float*)d_in[6];
  const float* bkc    = (const float*)d_in[7];
  const float* lns    = (const float*)d_in[8];
  const float* lnb    = (const float*)d_in[9];
  const float* w_in   = (const float*)d_in[10];
  const float* b_in   = (const float*)d_in[11];
  const float* w_gate = (const float*)d_in[12];
  const float* b_gate = (const float*)d_in[13];
  const float* w_out  = (const float*)d_in[14];
  const float* b_out  = (const float*)d_in[15];
  float* out = (float*)d_out;

  const int NSPLIT = 4;
  char* ws = (char*)d_ws;
  float* k_disp = (float*)ws;  ws += (size_t)BB * LL * DM * 4;
  float* qc     = (float*)ws;  ws += (size_t)BB * LL * DM * 4;
  float* o1     = (float*)ws;  ws += (size_t)BB * SS * DM * 4;
  float* o2     = (float*)ws;  ws += (size_t)BB * SS * DM * 4;
  float* kc     = (float*)ws;  ws += (size_t)BB * SS * DM * 4;
  __bf16* h_ws  = (__bf16*)ws; ws += (size_t)EE * 64 * DM * 2;
  __bf16* y_ws  = (__bf16*)ws; ws += (size_t)EE * 64 * DFF * 2;
  float* po     = (float*)ws;  ws += (size_t)BB * NH * NSPLIT * 128 * DH * 4;
  float* pmb    = (float*)ws;  ws += (size_t)BB * NH * NSPLIT * 128 * 4;
  float* plb    = (float*)ws;  ws += (size_t)BB * NH * NSPLIT * 128 * 4;

  dim3 blk(256);

  // 1) dispatch keys + combine queries: big GEMMs over x
  gemm_bias_kernel<<<dim3(DM / 128, (BB * LL) / 64), blk, 0, stream>>>(
      x, wkd, bkd, k_disp, BB * LL, DM, DM);
  gemm_bias_kernel<<<dim3(DM / 128, (BB * LL) / 64), blk, 0, stream>>>(
      x, wqc, bqc, qc, BB * LL, DM, DM);

  // 2) dispatch attention: slots (qbs=0) attend over L tokens; split-K over L
  attn_kernel<<<dim3(BB * NH * NSPLIT), blk, 0, stream>>>(
      slots, k_disp, x, (float*)nullptr, SS, LL, 0L, (long)LL * DM,
      (long)SS * DM, NSPLIT, po, pmb, plb);
  attn_merge_kernel<<<dim3(BB * NH), blk, 0, stream>>>(
      po, pmb, plb, o1, NSPLIT, (long)SS * DM);

  // 3) expert MLP: LN -> fused u/g + gating -> w_out + bias + residual
  ln_kernel<<<dim3(BB * SS), blk, 0, stream>>>(o1, lns, lnb, h_ws);
  mlp_ug_kernel<<<dim3(DFF / 64, EE), blk, 0, stream>>>(
      h_ws, w_in, b_in, w_gate, b_gate, y_ws);
  mlp_out_kernel<<<dim3(DM / 64, EE), blk, 0, stream>>>(
      y_ws, w_out, b_out, o1, o2);

  // 4) combine keys, then tokens attend over the 128 slots (fused epilogue)
  gemm_bias_kernel<<<dim3(DM / 128, (BB * SS) / 64), blk, 0, stream>>>(
      o2, wkc, bkc, kc, BB * SS, DM, DM);
  attn_kernel<<<dim3(BB * NH * (LL / 128)), blk, 0, stream>>>(
      qc, kc, o2, out, LL, SS, (long)LL * DM, (long)SS * DM, (long)LL * DM,
      1, (float*)nullptr, (float*)nullptr, (float*)nullptr);
}
